// HiMambaBlock_35038343201496
// MI455X (gfx1250) — matrix operand order repaired and checked by reading, not compiled, not run on hardware
//
#include <hip/hip_runtime.h>
#include <hip/hip_bf16.h>
#include <math.h>

typedef __attribute__((ext_vector_type(16))) __bf16 v16bf;
typedef __attribute__((ext_vector_type(8)))  float  v8f;

union Frag { unsigned u[8]; v16bf v; };

#define MROWS 16384   // B*H*W
#define CDIM  192
#define LDIM  4096
#define BDIM  4

__device__ __forceinline__ float sigf(float x)  { return 1.f / (1.f + __expf(-x)); }
__device__ __forceinline__ float siluf(float x) { return x * sigf(x); }
__device__ __forceinline__ float softplusf(float x) { return (x > 20.f) ? x : log1pf(__expf(x)); }

// sequence position -> raster index for the 4 scan directions (involutions)
__device__ __forceinline__ int seqmap(int d, int t) {
  if (d == 0) return t;
  if (d == 1) return 4095 - t;
  if (d == 2) return ((t & 63) << 6) | (t >> 6);
  int u = 4095 - t; return ((u & 63) << 6) | (u >> 6);
}

// async 16B global->LDS copy (CDNA5 GLOBAL_LOAD_ASYNC_TO_LDS_B128, ASYNCcnt)
__device__ __forceinline__ void async_copy_b128(unsigned lds_off, const void* gaddr) {
  asm volatile("global_load_async_to_lds_b128 %0, %1, off"
               :: "v"(lds_off), "v"((unsigned long long)(size_t)gaddr)
               : "memory");
}
__device__ __forceinline__ void wait_asynccnt0() {
  asm volatile("s_wait_asynccnt 0x0" ::: "memory");
}

// ---------------------------------------------------------------------------
// bf16 WMMA GEMM: C[M,N] = A[M,K](bf16,row-major) x W[K,N](bf16,row-major)
// Block: 128 threads (4 waves). Block tile 64x64, K-stage 64.
// A tile staged via async global->LDS DMA; B tile staged transposed.
// Wave w: rows [w*16, w*16+16), all 64 cols -> 4 accumulators, 8 WMMA/stage.
// MODE: 0 = store f32, 1 = accumulate f32, 2 = store bf16.
// Requires M%64==0, N%64==0, K%64==0.
// ---------------------------------------------------------------------------
template <int MODE>
__global__ __launch_bounds__(128)
void gemm_bf16_wmma(const __bf16* __restrict__ A, int lda,
                    const __bf16* __restrict__ Wt, int ldw,
                    float* __restrict__ Cf, __bf16* __restrict__ Cb,
                    int ldc, int K)
{
  __shared__ __bf16 As[64][64];   // [m][k]
  __shared__ __bf16 Bs[64][64];   // [n][k] (transposed stage)
  const int tid  = threadIdx.x;
  const int wave = tid >> 5;
  const int lane = tid & 31;
  const int hi   = lane >> 4;
  const int lo   = lane & 15;
  const int n0   = blockIdx.x * 64;
  const int m0   = blockIdx.y * 64;

  v8f acc[4];
#pragma unroll
  for (int nt = 0; nt < 4; ++nt) acc[nt] = (v8f){0.f,0.f,0.f,0.f,0.f,0.f,0.f,0.f};

  for (int kk = 0; kk < K; kk += 64) {
    // prefetch next K-stage tiles into cache while this stage computes
    if (kk + 64 < K) {
      __builtin_prefetch(A  + (size_t)(m0 + (tid & 63)) * lda + kk + 64, 0, 1);
      __builtin_prefetch(Wt + (size_t)(kk + 64 + (tid & 63)) * ldw + n0, 0, 1);
    }
    // stage A via async DMA: 64x64 bf16 = 512 x 16B chunks, 4 per thread
#pragma unroll
    for (int i = 0; i < 4; ++i) {
      int chunk = tid + i * 128;
      int row = chunk >> 3, q = chunk & 7;
      async_copy_b128((unsigned)(size_t)&As[row][q * 8],
                      A + (size_t)(m0 + row) * lda + kk + q * 8);
    }
    // stage B transposed: coalesced uint4 reads along N, scatter to Bs[n][k]
#pragma unroll
    for (int i = 0; i < 4; ++i) {
      int chunk = tid + i * 128;
      int k = chunk >> 3, nq = chunk & 7;
      uint4 v = *(const uint4*)(Wt + (size_t)(kk + k) * ldw + n0 + nq * 8);
      const __bf16* pv = (const __bf16*)&v;
#pragma unroll
      for (int j = 0; j < 8; ++j) Bs[nq * 8 + j][k] = pv[j];
    }
    wait_asynccnt0();
    __syncthreads();

#pragma unroll
    for (int k2 = 0; k2 < 64; k2 += 32) {
      Frag fa;
      const unsigned* rA = (const unsigned*)(&As[wave * 16 + lo][k2]);
#pragma unroll
      for (int q = 0; q < 4; ++q) { fa.u[q] = rA[hi * 4 + q]; fa.u[4 + q] = rA[hi * 4 + 8 + q]; }
#pragma unroll
      for (int nt = 0; nt < 4; ++nt) {
        Frag fb;
        const unsigned* rB = (const unsigned*)(&Bs[nt * 16 + lo][k2]);
#pragma unroll
        for (int q = 0; q < 4; ++q) { fb.u[q] = rB[hi * 4 + q]; fb.u[4 + q] = rB[hi * 4 + 8 + q]; }
        acc[nt] = __builtin_amdgcn_wmma_f32_16x16x32_bf16(
            false, fa.v, false, fb.v, (short)0, acc[nt], false, false);
      }
    }
    __syncthreads();
  }

  const int row0 = m0 + wave * 16 + hi * 8;
#pragma unroll
  for (int nt = 0; nt < 4; ++nt) {
    int col = n0 + nt * 16 + lo;
#pragma unroll
    for (int r = 0; r < 8; ++r) {
      size_t o = (size_t)(row0 + r) * ldc + col;
      if (MODE == 1)      Cf[o] += acc[nt][r];
      else if (MODE == 2) Cb[o]  = (__bf16)acc[nt][r];
      else                Cf[o]  = acc[nt][r];
    }
  }
}

// ---------------------------------------------------------------------------
// LayerNorm over C: NCHW input -> NHWC bf16 rows [M, C]
// ---------------------------------------------------------------------------
__global__ __launch_bounds__(192)
void ln_kernel(const float* __restrict__ x, const float* __restrict__ g,
               const float* __restrict__ bb, __bf16* __restrict__ xnb)
{
  __shared__ float s1[192], s2[192];
  int row = blockIdx.x;
  int c = threadIdx.x;
  int b = row >> 12, hw = row & 4095, h = hw >> 6, w = hw & 63;
  float v = x[(((size_t)b * 192 + c) * 64 + h) * 64 + w];
  s1[c] = v; s2[c] = v * v;
  __syncthreads();
  for (int off = 96; off >= 3; off >>= 1) {
    if (c < off) { s1[c] += s1[c + off]; s2[c] += s2[c + off]; }
    __syncthreads();
  }
  float m   = (s1[0] + s1[1] + s1[2]) * (1.f / 192.f);
  float var = (s2[0] + s2[1] + s2[2]) * (1.f / 192.f) - m * m;
  float y = (v - m) * rsqrtf(var + 1e-5f) * g[c] + bb[c];
  xnb[(size_t)row * 192 + c] = (__bf16)y;
}

// f32 -> bf16 pack with zero padding and optional transpose
__global__ void pack_kernel(const float* __restrict__ src, int srows, int scols,
                            __bf16* __restrict__ dst, int drows, int dcols, int transp)
{
  int i = blockIdx.x * blockDim.x + threadIdx.x;
  if (i >= drows * dcols) return;
  int r = i / dcols, c = i % dcols;
  float v = 0.f;
  if (transp) { if (c < srows && r < scols) v = src[(size_t)c * scols + r]; }
  else        { if (r < srows && c < scols) v = src[(size_t)r * scols + c]; }
  dst[i] = (__bf16)v;
}

// region causal depthwise conv (K=4) along direction-d sequence + SiLU
__global__ void conv_region_kernel(int d, const float* __restrict__ xz,
                                   const float* __restrict__ cw, const float* __restrict__ cb,
                                   float* __restrict__ xcf, __bf16* __restrict__ xcb)
{
  int i = blockIdx.x * blockDim.x + threadIdx.x;
  if (i >= MROWS * CDIM) return;
  int c = i % 192;
  int rt = i / 192;                 // b*4096 + t (sequence order)
  int t = rt & 4095, b = rt >> 12;
  float acc = cb[d * 192 + c];
#pragma unroll
  for (int k = 0; k < 4; ++k) {
    int s = t - 3 + k;
    if (s >= 0) {
      int ridx = seqmap(d, s);
      acc += xz[((size_t)(b * 4096 + ridx)) * 384 + c] * cw[d * 768 + c * 4 + k];
    }
  }
  acc = siluf(acc);
  xcf[(size_t)rt * 192 + c] = acc;
  xcb[(size_t)rt * 192 + c] = (__bf16)acc;
}

// dt_raw cols [0..11] of dbl[:,64] -> zero-padded bf16 [M,64]
__global__ void dtpack_kernel(const float* __restrict__ dbl, __bf16* __restrict__ dtp)
{
  int i = blockIdx.x * blockDim.x + threadIdx.x;
  if (i >= MROWS * 64) return;
  int col = i & 63, row = i >> 6;
  dtp[i] = (col < 12) ? (__bf16)dbl[(size_t)row * 64 + col] : (__bf16)0.0f;
}

// region selective scan: one block per batch, one thread per channel, N=16
__global__ __launch_bounds__(192)
void scan_region_kernel(int d, const float* __restrict__ xcf, const float* __restrict__ dbl,
                        const float* __restrict__ dtf, const float* __restrict__ xz,
                        const float* __restrict__ dtb, const float* __restrict__ alog,
                        const float* __restrict__ Dk, __bf16* __restrict__ yg)
{
  __shared__ float sB[16], sC[16];
  int b = blockIdx.x, c = threadIdx.x;
  float A[16], h[16];
#pragma unroll
  for (int n = 0; n < 16; ++n) { A[n] = -__expf(alog[(size_t)d * 3072 + c * 16 + n]); h[n] = 0.f; }
  float dbias = dtb[d * 192 + c];
  float Dc    = Dk[d * 192 + c];
  for (int t = 0; t < 4096; ++t) {
    size_t row = (size_t)b * 4096 + t;
    __syncthreads();
    if (c < 16)      sB[c]      = dbl[row * 64 + 12 + c];
    else if (c < 32) sC[c - 16] = dbl[row * 64 + 28 + (c - 16)];
    __syncthreads();
    float dt = softplusf(dtf[row * 192 + c] + dbias);
    float xv = xcf[row * 192 + c];
    float dx = dt * xv, y = 0.f;
#pragma unroll
    for (int n = 0; n < 16; ++n) {
      h[n] = __expf(dt * A[n]) * h[n] + dx * sB[n];
      y += h[n] * sC[n];
    }
    y += xv * Dc;
    int ridx = seqmap(d, t);
    size_t zr = (size_t)b * 4096 + ridx;
    float z = xz[zr * 384 + 192 + c];
    yg[zr * 192 + c] = (__bf16)(y * siluf(z));     // scatter to raster order
  }
}

// local windowed conv (K=2) within 8x8 windows + SiLU
__global__ void conv_local_kernel(const float* __restrict__ xz,
                                  const float* __restrict__ cw, const float* __restrict__ cb,
                                  float* __restrict__ xcf, __bf16* __restrict__ xcb)
{
  int i = blockIdx.x * blockDim.x + threadIdx.x;
  if (i >= MROWS * CDIM) return;
  int c = i % 192;
  int rs = i / 192;                 // wi*64 + t
  int t = rs & 63, wi = rs >> 6;
  int b = wi >> 6, wh = (wi >> 3) & 7, ww = wi & 7;
  auto rasterAt = [&](int tt) {
    int h = wh * 8 + (tt >> 3), w = ww * 8 + (tt & 7);
    return b * 4096 + h * 64 + w;
  };
  float acc = cb[c] + xz[(size_t)rasterAt(t) * 384 + c] * cw[c * 2 + 1];
  if (t > 0) acc += xz[(size_t)rasterAt(t - 1) * 384 + c] * cw[c * 2 + 0];
  acc = siluf(acc);
  xcf[(size_t)rs * 192 + c] = acc;
  xcb[(size_t)rs * 192 + c] = (__bf16)acc;
}

// local scan: one block per window (256), one thread per channel, N=8, L=64
__global__ __launch_bounds__(192)
void scan_local_kernel(const float* __restrict__ xcf, const float* __restrict__ dbl,
                       const float* __restrict__ dtf, const float* __restrict__ xz,
                       const float* __restrict__ dtb, const float* __restrict__ alog,
                       const float* __restrict__ Dk, __bf16* __restrict__ yg)
{
  __shared__ float sB[8], sC[8];
  int wi = blockIdx.x, c = threadIdx.x;
  int b = wi >> 6, wh = (wi >> 3) & 7, ww = wi & 7;
  float A[8], h[8];
#pragma unroll
  for (int n = 0; n < 8; ++n) { A[n] = -__expf(alog[c * 8 + n]); h[n] = 0.f; }
  float dbias = dtb[c], Dc = Dk[c];
  for (int t = 0; t < 64; ++t) {
    size_t row = (size_t)wi * 64 + t;
    __syncthreads();
    if (c < 8)       sB[c]     = dbl[row * 64 + 12 + c];
    else if (c < 16) sC[c - 8] = dbl[row * 64 + 20 + (c - 8)];
    __syncthreads();
    float dt = softplusf(dtf[row * 192 + c] + dbias);
    float xv = xcf[row * 192 + c];
    float dx = dt * xv, y = 0.f;
#pragma unroll
    for (int n = 0; n < 8; ++n) { h[n] = __expf(dt * A[n]) * h[n] + dx * sB[n]; y += h[n] * sC[n]; }
    y += xv * Dc;
    int hh = wh * 8 + (t >> 3), w2 = ww * 8 + (t & 7);
    size_t zr = (size_t)b * 4096 + hh * 64 + w2;
    float z = xz[zr * 384 + 192 + c];
    yg[zr * 192 + c] = (__bf16)(y * siluf(z));
  }
}

__global__ void fuse_kernel(const float* __restrict__ reg, const float* __restrict__ xl,
                            const float* __restrict__ scale,
                            float* __restrict__ ff, __bf16* __restrict__ fb)
{
  int i = blockIdx.x * blockDim.x + threadIdx.x;
  if (i >= MROWS * CDIM) return;
  float v = reg[i] * (0.25f * scale[0]) + xl[i];
  ff[i] = v; fb[i] = (__bf16)v;
}

// 3x3 depthwise conv (c1 bias folded at tap reads) + gated split h1*sigmoid(h2)
__global__ void dwgate_kernel(const __bf16* __restrict__ h1, const float* __restrict__ c1b,
                              const float* __restrict__ dww, const float* __restrict__ dwb,
                              __bf16* __restrict__ gated)
{
  int i = blockIdx.x * blockDim.x + threadIdx.x;
  if (i >= MROWS * 384) return;
  int ch = i % 384, row = i / 384;
  int b = row >> 12, hw = row & 4095, h = hw >> 6, w = hw & 63;
  float a1 = dwb[ch], a2 = dwb[ch + 384];
  float b1 = c1b[ch], b2 = c1b[ch + 384];
#pragma unroll
  for (int dy = -1; dy <= 1; ++dy)
#pragma unroll
    for (int dx = -1; dx <= 1; ++dx) {
      int hh = h + dy, ww2 = w + dx;
      if (hh >= 0 && hh < 64 && ww2 >= 0 && ww2 < 64) {
        size_t r2 = (size_t)b * 4096 + hh * 64 + ww2;
        int ti = (dy + 1) * 3 + (dx + 1);
        a1 += ((float)h1[r2 * 768 + ch]       + b1) * dww[ch * 9 + ti];
        a2 += ((float)h1[r2 * 768 + ch + 384] + b2) * dww[(ch + 384) * 9 + ti];
      }
    }
  gated[(size_t)row * 384 + ch] = (__bf16)(a1 * sigf(a2));
}

__global__ void final_kernel(const float* __restrict__ x, const float* __restrict__ ff,
                             const float* __restrict__ sg, const float* __restrict__ c2b,
                             const float* __restrict__ gamma, float* __restrict__ out)
{
  int i = blockIdx.x * blockDim.x + threadIdx.x;
  if (i >= BDIM * CDIM * LDIM) return;
  int w = i & 63, h = (i >> 6) & 63;
  int c = (i >> 12) % 192;
  int b = i / (192 * 4096);
  size_t row = (size_t)b * 4096 + h * 64 + w;
  out[i] = x[i] + gamma[0] * (ff[row * 192 + c] + sg[row * 192 + c] + c2b[c]);
}

// ---------------------------------------------------------------------------
extern "C" void kernel_launch(void* const* d_in, const int* in_sizes, int n_in,
                              void* d_out, int out_size, void* d_ws, size_t ws_size,
                              hipStream_t stream)
{
  (void)in_sizes; (void)n_in; (void)out_size; (void)ws_size;
  const float* x        = (const float*)d_in[0];
  const float* r_in_w   = (const float*)d_in[1];
  const float* r_conv_w = (const float*)d_in[2];
  const float* r_conv_b = (const float*)d_in[3];
  const float* r_xproj  = (const float*)d_in[4];
  const float* r_dt_w   = (const float*)d_in[5];
  const float* r_dt_b   = (const float*)d_in[6];
  const float* r_Alog   = (const float*)d_in[7];
  const float* r_D      = (const float*)d_in[8];
  const float* r_out_w  = (const float*)d_in[9];
  const float* scale    = (const float*)d_in[10];
  const float* l_in_w   = (const float*)d_in[11];
  const float* l_conv_w = (const float*)d_in[12];
  const float* l_conv_b = (const float*)d_in[13];
  const float* l_xproj  = (const float*)d_in[14];
  const float* l_dt_w   = (const float*)d_in[15];
  const float* l_dt_b   = (const float*)d_in[16];
  const float* l_Alog   = (const float*)d_in[17];
  const float* l_D      = (const float*)d_in[18];
  const float* l_out_w  = (const float*)d_in[19];
  const float* ln_g     = (const float*)d_in[20];
  const float* ln_b     = (const float*)d_in[21];
  const float* c1_w     = (const float*)d_in[22];
  const float* c1_b     = (const float*)d_in[23];
  const float* dw_w     = (const float*)d_in[24];
  const float* dw_b     = (const float*)d_in[25];
  const float* c2_w     = (const float*)d_in[26];
  const float* c2_b     = (const float*)d_in[27];
  const float* gamma    = (const float*)d_in[28];

  char* ws = (char*)d_ws;
  size_t off = 0;
  auto alloc = [&](size_t bytes) { size_t o = off; off += (bytes + 255) & ~(size_t)255; return o; };
  const size_t M = MROWS;

  size_t oXN   = alloc(M * 192 * 2);   // LN'd tokens, bf16 NHWC rows
  size_t oREG  = alloc(M * 192 * 4);   // region out accumulator (f32)
  size_t oXL   = alloc(M * 192 * 4);   // local out (f32)
  size_t oFF   = alloc(M * 192 * 4);   // fused f32
  size_t oFB   = alloc(M * 192 * 2);   // fused bf16
  size_t oSG   = alloc(M * 192 * 4);   // sgfn c2 out (f32)
  size_t oXZ   = alloc(M * 384 * 4);   // xz f32 (also aliased as SGFN h1 bf16 [M,768])
  size_t oXCF  = alloc(M * 192 * 4);   // xc f32 (also aliased as gated bf16 [M,384])
  size_t oXCB  = alloc(M * 192 * 2);   // xc bf16
  size_t oDBL  = alloc(M * 64 * 4);    // xproj out (f32, padded N=64)
  size_t oDTP  = alloc(M * 64 * 2);    // dt_raw packed bf16 (K padded 12->64)
  size_t oDTF  = alloc(M * 192 * 4);   // dt pre-softplus (f32)
  size_t oYG   = alloc(M * 192 * 2);   // gated y bf16 (raster order)
  size_t oWRIN = alloc(4 * 192 * 384 * 2);
  size_t oWRXP = alloc(4 * 192 * 64 * 2);
  size_t oWRDT = alloc(4 * 64 * 192 * 2);
  size_t oWROUT= alloc(4 * 192 * 192 * 2);
  size_t oWLIN = alloc(192 * 384 * 2);
  size_t oWLXP = alloc(192 * 64 * 2);
  size_t oWLDT = alloc(64 * 192 * 2);
  size_t oWLOUT= alloc(192 * 192 * 2);
  size_t oWC1  = alloc(192 * 768 * 2); // c1_w transposed [K=192,N=768]
  size_t oWC2  = alloc(384 * 192 * 2); // c2_w transposed [K=384,N=192]

  auto pack = [&](const float* src, int sr, int sc, size_t dOff, int dr, int dc, int tr) {
    int tot = dr * dc;
    pack_kernel<<<(tot + 255) / 256, 256, 0, stream>>>(src, sr, sc, (__bf16*)(ws + dOff), dr, dc, tr);
  };
  auto gemm = [&](size_t aOff, int lda, size_t wOff, int ldw, size_t cOff, int ldc,
                  int Mr, int Nr, int Kr, int mode) {
    dim3 g(Nr / 64, Mr / 64);
    const __bf16* Ap = (const __bf16*)(ws + aOff);
    const __bf16* Wp = (const __bf16*)(ws + wOff);
    float* Cf = (float*)(ws + cOff);
    __bf16* Cb = (__bf16*)(ws + cOff);
    if (mode == 0)      gemm_bf16_wmma<0><<<g, 128, 0, stream>>>(Ap, lda, Wp, ldw, Cf, Cb, ldc, Kr);
    else if (mode == 1) gemm_bf16_wmma<1><<<g, 128, 0, stream>>>(Ap, lda, Wp, ldw, Cf, Cb, ldc, Kr);
    else                gemm_bf16_wmma<2><<<g, 128, 0, stream>>>(Ap, lda, Wp, ldw, Cf, Cb, ldc, Kr);
  };

  // ---- weight packing (f32 -> bf16, with K/N padding and transposes) ----
  for (int d = 0; d < 4; ++d) {
    pack(r_in_w  + (size_t)d * 192 * 384, 192, 384, oWRIN + (size_t)d * 192 * 384 * 2, 192, 384, 0);
    pack(r_xproj + (size_t)d * 192 * 44,  192, 44,  oWRXP + (size_t)d * 192 * 64 * 2,  192, 64,  0);
    pack(r_dt_w  + (size_t)d * 12 * 192,  12,  192, oWRDT + (size_t)d * 64 * 192 * 2,  64,  192, 0);
    pack(r_out_w + (size_t)d * 192 * 192, 192, 192, oWROUT+ (size_t)d * 192 * 192 * 2, 192, 192, 0);
  }
  pack(l_in_w,  192, 384, oWLIN, 192, 384, 0);
  pack(l_xproj, 192, 44,  oWLXP, 192, 64,  0);
  pack(l_dt_w,  12,  192, oWLDT, 64,  192, 0);
  pack(l_out_w, 192, 192, oWLOUT,192, 192, 0);
  pack(c1_w, 768, 192, oWC1, 192, 768, 1);
  pack(c2_w, 192, 384, oWC2, 384, 192, 1);

  // ---- layernorm -> bf16 token rows ----
  ln_kernel<<<(int)M, 192, 0, stream>>>(x, ln_g, ln_b, (__bf16*)(ws + oXN));

  const int ME = MROWS * CDIM;

  // ---- region 4-direction pipeline (buffers reused per direction) ----
  for (int d = 0; d < 4; ++d) {
    gemm(oXN, 192, oWRIN + (size_t)d * 192 * 384 * 2, 384, oXZ, 384, (int)M, 384, 192, 0);
    conv_region_kernel<<<(ME + 255) / 256, 256, 0, stream>>>(d, (const float*)(ws + oXZ),
        r_conv_w, r_conv_b, (float*)(ws + oXCF), (__bf16*)(ws + oXCB));
    gemm(oXCB, 192, oWRXP + (size_t)d * 192 * 64 * 2, 64, oDBL, 64, (int)M, 64, 192, 0);
    dtpack_kernel<<<((int)(M * 64) + 255) / 256, 256, 0, stream>>>((const float*)(ws + oDBL),
        (__bf16*)(ws + oDTP));
    gemm(oDTP, 64, oWRDT + (size_t)d * 64 * 192 * 2, 192, oDTF, 192, (int)M, 192, 64, 0);
    scan_region_kernel<<<4, 192, 0, stream>>>(d, (const float*)(ws + oXCF),
        (const float*)(ws + oDBL), (const float*)(ws + oDTF), (const float*)(ws + oXZ),
        r_dt_b, r_Alog, r_D, (__bf16*)(ws + oYG));
    gemm(oYG, 192, oWROUT + (size_t)d * 192 * 192 * 2, 192, oREG, 192, (int)M, 192, 192,
         d == 0 ? 0 : 1);
  }

  // ---- local windowed pipeline ----
  gemm(oXN, 192, oWLIN, 384, oXZ, 384, (int)M, 384, 192, 0);
  conv_local_kernel<<<(ME + 255) / 256, 256, 0, stream>>>((const float*)(ws + oXZ),
      l_conv_w, l_conv_b, (float*)(ws + oXCF), (__bf16*)(ws + oXCB));
  gemm(oXCB, 192, oWLXP, 64, oDBL, 64, (int)M, 64, 192, 0);
  dtpack_kernel<<<((int)(M * 64) + 255) / 256, 256, 0, stream>>>((const float*)(ws + oDBL),
      (__bf16*)(ws + oDTP));
  gemm(oDTP, 64, oWLDT, 192, oDTF, 192, (int)M, 192, 64, 0);
  scan_local_kernel<<<256, 192, 0, stream>>>((const float*)(ws + oXCF),
      (const float*)(ws + oDBL), (const float*)(ws + oDTF), (const float*)(ws + oXZ),
      l_dt_b, l_Alog, l_D, (__bf16*)(ws + oYG));
  gemm(oYG, 192, oWLOUT, 192, oXL, 192, (int)M, 192, 192, 0);

  // ---- fuse + SGFN + residual ----
  fuse_kernel<<<(ME + 255) / 256, 256, 0, stream>>>((const float*)(ws + oREG),
      (const float*)(ws + oXL), scale, (float*)(ws + oFF), (__bf16*)(ws + oFB));
  // h1 (bf16) written into the xz arena (same byte size: M*384*4 == M*768*2)
  gemm(oFB, 192, oWC1, 768, oXZ, 768, (int)M, 768, 192, 2);
  dwgate_kernel<<<((int)(M * 384) + 255) / 256, 256, 0, stream>>>((const __bf16*)(ws + oXZ),
      c1_b, dw_w, dw_b, (__bf16*)(ws + oXCF));   // gated bf16 aliases xc arena
  gemm(oXCF, 384, oWC2, 192, oSG, 192, (int)M, 192, 384, 0);
  final_kernel<<<(ME + 255) / 256, 256, 0, stream>>>(x, (const float*)(ws + oFF),
      (const float*)(ws + oSG), c2_b, gamma, (float*)d_out);
}
